// FFF_61838939128374
// MI455X (gfx1250) — compile-verified, matrix operand order checked by reading, not run on hardware
//
#include <hip/hip_runtime.h>
#include <hip/hip_bf16.h>
#include <cstdint>

typedef __attribute__((ext_vector_type(16))) _Float16 v16h;
typedef __attribute__((ext_vector_type(8)))  float    v8f;
typedef int v4i __attribute__((vector_size(16)));   // matches async-LDS builtin param type

#define IN_F   1024
#define LW     64
#define OUT_F  512
#define NLEAF  64
#define NNODE  63
#define DEPTHT 6
#define BTOT   4096
#define MTILES (BTOT/16)    // 256 16-row fragments tiles of x
#define KS1    (IN_F/32)    // 32 k-steps for K=1024
#define KS2    (LW/32)      // 2 k-steps for K=64
#define NT1    (LW/16)      // 4 n-tiles of h
#define NT2    (OUT_F/16)   // 32 n-tiles of out
#define NBLK   128          // main-kernel blocks: 32 batch rows each

#define WMMA_F16(a,b,c) __builtin_amdgcn_wmma_f32_16x16x32_f16(false,(a),false,(b),(short)0,(c),false,false)

#if defined(__HIP_DEVICE_COMPILE__) && __has_builtin(__builtin_amdgcn_global_load_async_to_lds_b128) && __has_builtin(__builtin_amdgcn_s_wait_asynccnt)
#define HAVE_ASYNC_LDS 1
#else
#define HAVE_ASYNC_LDS 0
#endif

union FragU { uint4 q[2]; v16h h; };

// Load a pre-swizzled fragment (1 KB: part0 = lanes*16B, part1 = +512B).
static __device__ __forceinline__ v16h ld_frag(const uint4* fb, int lane) {
  FragU f;
  f.q[0] = fb[lane];
  f.q[1] = fb[32 + lane];
  return f.h;
}

// Build an A-matrix fragment (16xK tile, f16) from a row-major LDS buffer.
// ISA 7.12.2: lane = M (two halves), VGPR i: g=i>>2, j=i&3, K = 16g + 8h + 2j (+pos).
static __device__ __forceinline__ v16h ld_a_rowmajor_lds(const _Float16* buf, int stride,
                                                         int kbase, int lane) {
  const int m = lane & 15, hh = lane >> 4;
  union { unsigned int u[8]; v16h h; } f;
#pragma unroll
  for (int i = 0; i < 8; ++i) {
    const int g = i >> 2, j = i & 3;
    const int k = kbase + 16*g + 8*hh + 2*j;
    f.u[i] = *(const unsigned int*)(buf + m*stride + k);   // pairs (k,k+1) contiguous
  }
  return f.h;
}

static __device__ __forceinline__ float fast_sigmoid(float x) {
  const float d = 1.f + __expf(-x);
#if defined(__HIP_DEVICE_COMPILE__) && __has_builtin(__builtin_amdgcn_rcpf)
  return __builtin_amdgcn_rcpf(d);      // v_rcp_f32, avoids IEEE div expansion
#else
  return 1.f / d;
#endif
}

static __device__ __forceinline__ unsigned short h16(float v) {
  union { _Float16 h; unsigned short s; } c; c.h = (_Float16)v; return c.s;
}

// ---- Pack x (f32 row-major) into A-type WMMA fragments (f16). ----
__global__ void FFF_pack_x(const float* __restrict__ x, uint4* __restrict__ dst) {
  const long t = (long)blockIdx.x * blockDim.x + threadIdx.x;   // one thread per (frag, lane)
  const int  lane = (int)(t & 31);
  const long frag = t >> 5;
  if (frag >= (long)MTILES * KS1) return;
  const int  ks   = (int)(frag & (KS1 - 1));
  const long tile = frag >> 5;
  const int  hh = lane >> 4, m = lane & 15;
  unsigned int w[8];
#pragma unroll
  for (int i = 0; i < 8; ++i) {
    const int g = i >> 2, j = i & 3;
    const int k = ks*32 + 16*g + 8*hh + 2*j;
    const float v0 = x[(tile*16 + m)*IN_F + k];
    const float v1 = x[(tile*16 + m)*IN_F + k + 1];
    w[i] = (unsigned int)h16(v0) | ((unsigned int)h16(v1) << 16);
  }
  uint4* o = dst + frag*64;
  o[lane]      = make_uint4(w[0], w[1], w[2], w[3]);
  o[32 + lane] = make_uint4(w[4], w[5], w[6], w[7]);
}

// ---- Generic pack of f32 weights into B-type WMMA fragments (f16). ----
// B[k][n] = src[leaf*leafStride + k*kStride + n*nStride], zero-padded past limits.
__global__ void FFF_pack_b(const float* __restrict__ src, uint4* __restrict__ dst,
                           int ntCount, int ksCount, long nFrags,
                           long leafStride, int kStride, int nStride,
                           int nLimit, int kLimit) {
  const long t = (long)blockIdx.x * blockDim.x + threadIdx.x;
  const int  lane = (int)(t & 31);
  const long frag = t >> 5;
  if (frag >= nFrags) return;
  const int  perLeaf = ntCount * ksCount;
  const long leaf = frag / perLeaf;
  const int  rem  = (int)(frag - leaf*perLeaf);
  const int  nt   = rem / ksCount;
  const int  ks   = rem - nt*ksCount;
  const int  hh   = lane >> 4;
  const int  n    = nt*16 + (lane & 15);
  unsigned int w[8];
#pragma unroll
  for (int i = 0; i < 8; ++i) {
    const int g = i >> 2, j = i & 3;
    const int k0 = ks*32 + 16*g + 8*hh + 2*j;
    float v0 = 0.f, v1 = 0.f;
    if (n < nLimit) {
      if (k0     < kLimit) v0 = src[leaf*leafStride + (long)k0    *kStride + (long)n*nStride];
      if (k0 + 1 < kLimit) v1 = src[leaf*leafStride + (long)(k0+1)*kStride + (long)n*nStride];
    }
    w[i] = (unsigned int)h16(v0) | ((unsigned int)h16(v1) << 16);
  }
  uint4* o = dst + frag*64;
  o[lane]      = make_uint4(w[0], w[1], w[2], w[3]);
  o[32 + lane] = make_uint4(w[4], w[5], w[6], w[7]);
}

// ---- Main fused FFF kernel: 1 block = 32 batch rows (2 M-tiles), 8 waves. ----
// Every B fragment is loaded once per block and reused for both M-tiles.
__global__ __launch_bounds__(256) void FFF_main(
    const uint4* __restrict__ xp,
    const uint4* __restrict__ w1p,
    const uint4* __restrict__ w2p,
    const uint4* __restrict__ npk,
    const uint4* __restrict__ b2p,
    const float* __restrict__ node_biases,
    const float* __restrict__ b1s,
    float* __restrict__ out)
{
  __shared__ uint4    xfr[2][KS1 * 64];         // 64 KB: both M-tiles' A fragments
  __shared__ float    sig[32][64];              // 8 KB: sigmoid of routing logits
  __shared__ float    mixf[32][64];             // 8 KB: per-leaf mixture (f32)
  __shared__ _Float16 mix16[32][72];            // 4.5 KB: mixture (f16, padded rows)
  __shared__ _Float16 hbuf[2][32][72];          // 9 KB: h' for the 2 in-flight leaves

  const int tid  = threadIdx.x;
  const int wave = tid >> 5;
  const int lane = tid & 31;
  const int bt   = blockIdx.x;                  // 32-row block index
  const int ln   = lane & 15, hh = lane >> 4;

  // Stage both M-tiles' packed x fragments into LDS (async DMA path if available).
  {
    const uint4* xtile = xp + (long)bt * 2 * (KS1 * 64);
#if HAVE_ASYNC_LDS
    uint4* l = &xfr[0][0];
    for (int i = tid; i < 2 * KS1 * 64; i += 256) {
      __builtin_amdgcn_global_load_async_to_lds_b128(
          (v4i*)(xtile + i), (v4i*)(l + i), 0, 0);
    }
    __builtin_amdgcn_s_wait_asynccnt(0);
#else
    for (int i = tid; i < 2 * KS1 * 64; i += 256) xfr[0][i] = xtile[i];
#endif
  }
  __syncthreads();

  // ---- Routing: logits(32x64) = x @ node_w^T via WMMA (8 tiles over 8 waves) ----
  {
    const int mt = wave >> 2;
    const int nt = wave & 3;
    v8f c = {};
#pragma unroll 4
    for (int ks = 0; ks < KS1; ++ks) {
      v16h a = ld_frag(&xfr[mt][ks*64], lane);
      v16h b = ld_frag(npk + ((long)nt*KS1 + ks)*64, lane);
      c = WMMA_F16(a, b, c);
    }
    const int node = nt*16 + ln;
    const float nb = (node < NNODE) ? node_biases[node] : 0.f;
#pragma unroll
    for (int r = 0; r < 8; ++r) {
      const int row = mt*16 + r + 8*hh;
      const float s = fast_sigmoid(c[r] + nb);
      if (node < NNODE) sig[row][node] = s;
    }
  }
  __syncthreads();

  // ---- Tree product -> per-leaf mixture (32 rows x 64 leaves) ----
  for (int e = tid; e < 32*NLEAF; e += 256) {
    const int m = e >> 6, leaf = e & (NLEAF - 1);
    float p = 1.f;
#pragma unroll
    for (int d = 0; d < DEPTHT; ++d) {
      const int g = leaf >> (5 - d);
      const int node = (1 << d) - 1 + (g >> 1);
      const float s = sig[m][node];
      p *= (g & 1) ? s : (1.f - s);
    }
    mixf[m][leaf]  = p;
    mix16[m][leaf] = (_Float16)p;
  }
  __syncthreads();

  v8f co[2][4];                                  // out accumulators: 2 M-tiles x 64 cols
  {
    v8f z = {};
#pragma unroll
    for (int mt = 0; mt < 2; ++mt)
#pragma unroll
      for (int q = 0; q < 4; ++q) co[mt][q] = z;
  }

  const int leafSel = wave >> 2;   // which leaf of the pair this wave computes h for
  const int nt1     = wave & 3;    // which 16-col tile of h

  for (int pair = 0; pair < NLEAF/2; ++pair) {
    const int leaf = pair*2 + leafSel;
    const uint4* w1base = w1p + ((long)(leaf*NT1 + nt1) * KS1) * 64;

    // Prefetch next pair's weight fragments into cache (global_prefetch_b8).
    if (pair + 1 < NLEAF/2) {
      const int leafN = (pair + 1)*2 + leafSel;
      __builtin_prefetch(w1p + ((long)(leafN*NT1 + nt1) * KS1) * 64 + lane*4, 0, 0);
      __builtin_prefetch(w2p + ((long)leafN * NT2 * KS2) * 64 + lane*4, 0, 0);
    }

    // GEMM1: h = relu(x @ w1[leaf] + b1) * mixture; one B load feeds both M-tiles.
    v8f c1[2] = {};
#pragma unroll 4
    for (int ks = 0; ks < KS1; ++ks) {
      v16h b  = ld_frag(w1base + ks*64, lane);
      v16h a0 = ld_frag(&xfr[0][ks*64], lane);
      v16h a1 = ld_frag(&xfr[1][ks*64], lane);
      c1[0] = WMMA_F16(a0, b, c1[0]);
      c1[1] = WMMA_F16(a1, b, c1[1]);
    }
    const float bb = b1s[leaf*LW + nt1*16 + ln];
#pragma unroll
    for (int mt = 0; mt < 2; ++mt) {
#pragma unroll
      for (int r = 0; r < 8; ++r) {
        const int row = mt*16 + r + 8*hh;
        float v = c1[mt][r] + bb;
        v = fmaxf(v, 0.f);
        v *= mixf[row][leaf];
        hbuf[leafSel][row][nt1*16 + ln] = (_Float16)v;
      }
    }
    __syncthreads();

    // GEMM2: out += h' @ w2[leaf]; wave owns 64 out cols; B reused across M-tiles.
#pragma unroll
    for (int li = 0; li < 2; ++li) {
      const int l2 = pair*2 + li;
      const uint4* w2base = w2p + ((long)l2 * NT2 * KS2) * 64;
#pragma unroll
      for (int ks = 0; ks < KS2; ++ks) {
        v16h a0 = ld_a_rowmajor_lds(&hbuf[li][0][0],  72, ks*32, lane);
        v16h a1 = ld_a_rowmajor_lds(&hbuf[li][16][0], 72, ks*32, lane);
#pragma unroll
        for (int q = 0; q < 4; ++q) {
          const int ont = wave*4 + q;
          v16h b = ld_frag(w2base + ((long)ont*KS2 + ks)*64, lane);
          co[0][q] = WMMA_F16(a0, b, co[0][q]);
          co[1][q] = WMMA_F16(a1, b, co[1][q]);
        }
      }
    }
    __syncthreads();
  }

  // b2 contribution: out += mixture(32x64) @ b2s(64x512), also via WMMA.
#pragma unroll
  for (int ks = 0; ks < KS2; ++ks) {
    v16h a0 = ld_a_rowmajor_lds(&mix16[0][0],  72, ks*32, lane);
    v16h a1 = ld_a_rowmajor_lds(&mix16[16][0], 72, ks*32, lane);
#pragma unroll
    for (int q = 0; q < 4; ++q) {
      v16h b = ld_frag(b2p + ((long)(wave*4 + q)*KS2 + ks)*64, lane);
      co[0][q] = WMMA_F16(a0, b, co[0][q]);
      co[1][q] = WMMA_F16(a1, b, co[1][q]);
    }
  }

  // Store 32x512 f32 output tile.
#pragma unroll
  for (int mt = 0; mt < 2; ++mt) {
#pragma unroll
    for (int q = 0; q < 4; ++q) {
#pragma unroll
      for (int r = 0; r < 8; ++r) {
        const long row = (long)bt*32 + mt*16 + r + 8*hh;
        out[row*OUT_F + wave*64 + q*16 + ln] = co[mt][q][r];
      }
    }
  }
}

extern "C" void kernel_launch(void* const* d_in, const int* in_sizes, int n_in,
                              void* d_out, int out_size, void* d_ws, size_t ws_size,
                              hipStream_t stream) {
  (void)in_sizes; (void)n_in; (void)out_size; (void)ws_size;
  const float* x  = (const float*)d_in[0];
  const float* nw = (const float*)d_in[1];
  const float* nb = (const float*)d_in[2];
  const float* w1 = (const float*)d_in[3];
  const float* b1 = (const float*)d_in[4];
  const float* w2 = (const float*)d_in[5];
  const float* b2 = (const float*)d_in[6];
  float* out = (float*)d_out;

  char* ws = (char*)d_ws;                       // ~20.2 MB used
  uint4* xp  = (uint4*)(ws);                    // 8 MiB: 256*32 frags
  uint4* w1p = (uint4*)(ws + (8l  << 20));      // 8 MiB: 64*4*32 frags
  uint4* w2p = (uint4*)(ws + (16l << 20));      // 4 MiB: 64*32*2 frags
  uint4* npk = (uint4*)(ws + (20l << 20));      // 128 KiB: 4*32 frags
  uint4* b2p = (uint4*)(ws + (20l << 20) + (128l << 10)); // 64 KiB: 32*2 frags

  // Pack passes (f32 -> f16 WMMA fragment layout), all on `stream`.
  FFF_pack_x<<<1024, 256, 0, stream>>>(x, xp);                                   // 8192 frags
  FFF_pack_b<<<1024, 256, 0, stream>>>(w1, w1p, 4, 32, 8192,
                                       (long)IN_F*LW, LW, 1, LW, IN_F);          // w1s
  FFF_pack_b<<<512, 256, 0, stream>>>(w2, w2p, 32, 2, 4096,
                                      (long)LW*OUT_F, OUT_F, 1, OUT_F, LW);      // w2s
  FFF_pack_b<<<16, 256, 0, stream>>>(nw, npk, 4, 32, 128,
                                     0, 1, IN_F, NNODE, IN_F);                   // node_w^T (pad 63->64)
  FFF_pack_b<<<8, 256, 0, stream>>>(b2, b2p, 32, 2, 64,
                                    0, OUT_F, 1, OUT_F, NLEAF);                  // b2s as 64x512 B

  FFF_main<<<NBLK, 256, 0, stream>>>(xp, w1p, w2p, npk, b2p, nb, b1, out);
}
